// CrossAttentionBlock_85598698209468
// MI455X (gfx1250) — matrix operand order repaired and checked by reading, not compiled, not run on hardware
//
#include <hip/hip_runtime.h>

typedef __attribute__((ext_vector_type(16))) __bf16 v16bf;
typedef __attribute__((ext_vector_type(8)))  __bf16 v8bf;
typedef __attribute__((ext_vector_type(8)))  float  v8f;

#define C_IN   256
#define C_KEY  128
#define C_VAL  128
#define C_OUT  256
#define BATCH  4
#define HWN    4096
#define BN_EPS 1e-5f

#define KSTRIDE 136   // 32x128 K tile padded: +16B/row -> conflict-free b128 reads
#define VSTRIDE 40    // 128x32 V tile padded: +16B/row -> conflict-free b128 reads

__device__ __forceinline__ v8f wmma_bf16(v16bf a, v16bf b, v8f c) {
  // D = A(16x32) * B(32x16) + C, f32 accumulate
  return __builtin_amdgcn_wmma_f32_16x16x32_bf16(false, a, false, b, (short)0, c,
                                                 false, false);
}

// A-matrix fragment 16x32 (MxK), bf16, from row-major src with leading dim ld.
// ISA layout: lane m=lane&15, half=lane>>4; elems 0..7 -> K=half*8+e,
// elems 8..15 -> K=16+half*8+(e-8).
__device__ __forceinline__ v16bf load_a_frag(const __bf16* __restrict__ src,
                                             int m0, int k0, int ld, int lane) {
  int m  = m0 + (lane & 15);
  int hf = (lane >> 4) & 1;
  const __bf16* p = src + (size_t)m * ld + k0 + hf * 8;
  union { v16bf v; v8bf h[2]; } u;
  u.h[0] = *(const v8bf*)(p);        // K = k0+half*8 .. +7
  u.h[1] = *(const v8bf*)(p + 16);   // K = k0+16+half*8 .. +7
  return u.v;
}

// B-matrix fragment 32x16 (KxN), bf16. Source row-major with N as the row
// index: element (k,n) = src[(n0+n)*ld + k0 + k].
// ISA layout: lane n=lane&15, half=lane>>4; elems 0..7 -> K=half*8+e,
// elems 8..15 -> K=16+half*8+(e-8).
__device__ __forceinline__ v16bf load_b_frag(const __bf16* __restrict__ src,
                                             int n0, int k0, size_t ld, int lane) {
  int n  = n0 + (lane & 15);
  int hf = (lane >> 4) & 1;
  const __bf16* p = src + (size_t)n * ld + k0 + hf * 8;
  union { v16bf v; v8bf h[2]; } u;
  u.h[0] = *(const v8bf*)(p);
  u.h[1] = *(const v8bf*)(p + 16);
  return u.v;
}

// B-fragment straight from an LDS tile (row-major, padded stride)
__device__ __forceinline__ v16bf lds_b_frag(const __bf16* buf, int n0, int k0,
                                            int stride, int lane) {
  const __bf16* p = buf + (n0 + (lane & 15)) * stride + k0 + ((lane >> 4) & 1) * 8;
  union { v16bf v; v8bf h[2]; } u;
  u.h[0] = *(const v8bf*)(p);
  u.h[1] = *(const v8bf*)(p + 16);
  return u.v;
}

// ---------------------------------------------------------------------------
// Async global -> LDS copy (CDNA5 GLOBAL_LOAD_ASYNC_TO_LDS_B128, ASYNCcnt).
// Low 32 bits of a flat shared pointer are the LDS byte address (ISA 10.2).
// ---------------------------------------------------------------------------
#ifndef ATTN_USE_ASYNC
#define ATTN_USE_ASYNC 1
#endif

#if ATTN_USE_ASYNC
__device__ __forceinline__ void copy16_g2l(const __bf16* g, __bf16* l) {
  unsigned loff = (unsigned)(uintptr_t)(void*)l;
  asm volatile("global_load_async_to_lds_b128 %0, %1, off"
               :: "v"(loff), "v"(g) : "memory");
}
#define WAIT_ASYNC_NEXT_INFLIGHT() asm volatile("s_wait_asynccnt 8" ::: "memory")
#define WAIT_ASYNC_ALL()           asm volatile("s_wait_asynccnt 0" ::: "memory")
#else
__device__ __forceinline__ void copy16_g2l(const __bf16* g, __bf16* l) {
  *(v8bf*)l = *(const v8bf*)g;
}
#define WAIT_ASYNC_NEXT_INFLIGHT()
#define WAIT_ASYNC_ALL()
#endif

// Cooperative stage of one 32-key block: K tile 32x128, V tile 128x32 (bf16).
// 128 threads x 8 x 16B = 16KB. 8 async instructions per wave per stage.
__device__ __forceinline__ void stage_kv(const __bf16* __restrict__ Kb,
                                         const __bf16* __restrict__ Vb,
                                         int n0, __bf16* kbuf, __bf16* vbuf,
                                         int tid) {
#pragma unroll
  for (int i = 0; i < 4; ++i) {            // K: 32 rows x 256B
    int q = tid + (i << 7);
    int row = q >> 4, seg = q & 15;
    copy16_g2l(Kb + (size_t)(n0 + row) * C_KEY + seg * 8,
               kbuf + row * KSTRIDE + seg * 8);
  }
#pragma unroll
  for (int i = 0; i < 4; ++i) {            // V: 128 rows x 64B
    int q = tid + (i << 7);
    int cv = q >> 2, seg = q & 3;
    copy16_g2l(Vb + (size_t)cv * HWN + n0 + seg * 8,
               vbuf + cv * VSTRIDE + seg * 8);
  }
}

// ---------------------------------------------------------------------------
// Kernel 0: f32 -> bf16 copy (for wo)
// ---------------------------------------------------------------------------
__global__ void cvt_bf16_kernel(const float* __restrict__ src,
                                __bf16* __restrict__ dst, int n) {
  int i = blockIdx.x * blockDim.x + threadIdx.x;
  if (i < n) dst[i] = (__bf16)src[i];
}

// ---------------------------------------------------------------------------
// Kernel 1: Q/K/V 1x1-conv projections (+BN/ReLU for Q,K), output bf16.
// Q,Kt: [B, HW, C_KEY]   Vt: [B, C_VAL, HW]
// ---------------------------------------------------------------------------
__global__ __launch_bounds__(256) void qkv_kernel(
    const float* __restrict__ x_enc, const float* __restrict__ x_dec,
    const float* __restrict__ wk, const float* __restrict__ bk,
    const float* __restrict__ gk, const float* __restrict__ betak,
    const float* __restrict__ mk, const float* __restrict__ vk,
    const float* __restrict__ wq, const float* __restrict__ bq,
    const float* __restrict__ gq, const float* __restrict__ betaq,
    const float* __restrict__ mq, const float* __restrict__ vq,
    const float* __restrict__ wv, const float* __restrict__ bv,
    __bf16* __restrict__ Q, __bf16* __restrict__ Kt, __bf16* __restrict__ Vt) {
  int idx = blockIdx.x * blockDim.x + threadIdx.x;  // 0 .. B*HW-1
  int b  = idx >> 12;
  int hw = idx & (HWN - 1);
  int o  = blockIdx.y;        // output channel (0..127)
  int which = blockIdx.z;     // 0=query(x_dec), 1=key(x_enc), 2=value(x_enc)

  const float* x = (which == 0) ? x_dec : x_enc;
  const float* w = (which == 0) ? wq : ((which == 1) ? wk : wv);
  float acc = (which == 0) ? bq[o] : ((which == 1) ? bk[o] : bv[o]);

  const float* xb = x + (size_t)b * C_IN * HWN + hw;
  const float* wr = w + o * C_IN;
#pragma unroll 8
  for (int c = 0; c < C_IN; ++c)
    acc = fmaf(wr[c], xb[(size_t)c * HWN], acc);

  if (which == 0) {
    float inv = rsqrtf(vq[o] + BN_EPS);
    acc = (acc - mq[o]) * (gq[o] * inv) + betaq[o];
    acc = fmaxf(acc, 0.f);
    Q[((size_t)b * HWN + hw) * C_KEY + o] = (__bf16)acc;
  } else if (which == 1) {
    float inv = rsqrtf(vk[o] + BN_EPS);
    acc = (acc - mk[o]) * (gk[o] * inv) + betak[o];
    acc = fmaxf(acc, 0.f);
    Kt[((size_t)b * HWN + hw) * C_KEY + o] = (__bf16)acc;
  } else {
    Vt[((size_t)b * C_VAL + o) * HWN + hw] = (__bf16)acc;
  }
}

// ---------------------------------------------------------------------------
// Kernel 2: fused flash attention. 4 waves/block; each wave owns 16 query
// rows; the block shares double-buffered async-staged K/V tiles in LDS.
// Ctx: [B, HW, C_VAL] bf16
// ---------------------------------------------------------------------------
__global__ __launch_bounds__(128) void attn_kernel(
    const __bf16* __restrict__ Q, const __bf16* __restrict__ Kt,
    const __bf16* __restrict__ Vt, __bf16* __restrict__ Ctx) {
  const int tid   = threadIdx.x;
  const int lane  = tid & 31;
  const int wave  = tid >> 5;
  const int b     = blockIdx.y;
  const int m0    = blockIdx.x * 64 + wave * 16;
  const int mlane = lane & 15;
  const int hf    = (lane >> 4) & 1;

  const __bf16* Qb = Q  + (size_t)b * HWN * C_KEY;
  const __bf16* Kb = Kt + (size_t)b * HWN * C_KEY;
  const __bf16* Vb = Vt + (size_t)b * C_VAL * HWN;

  __shared__ __bf16 ldsK[2][32 * KSTRIDE];    // 2 x 8.5 KB
  __shared__ __bf16 ldsV[2][128 * VSTRIDE];   // 2 x 10 KB
  __shared__ __bf16 ldsP[4][16 * 32];         // per-wave P repack

  // Q tile 16x128 as four 16x32 A-fragments (persistent in VGPRs)
  v16bf qa[4];
#pragma unroll
  for (int kc = 0; kc < 4; ++kc)
    qa[kc] = load_a_frag(Qb, m0, kc * 32, C_KEY, lane);

  v8f acc[8];
#pragma unroll
  for (int t = 0; t < 8; ++t)
#pragma unroll
    for (int r = 0; r < 8; ++r) acc[t][r] = 0.f;

  // per-row softmax state; element r corresponds to query row r + 8*hf
  float mrow[8], lrow[8];
#pragma unroll
  for (int r = 0; r < 8; ++r) { mrow[r] = -3.0e38f; lrow[r] = 0.f; }

  __bf16* pbuf = ldsP[wave];
  const float scale = 0.088388347648318447f;  // 1/sqrt(128)
  const float LOG2E = 1.4426950408889634f;
  const int NB = HWN / 32;

  stage_kv(Kb, Vb, 0, ldsK[0], ldsV[0], tid);   // prologue: stage block 0

  for (int j = 0; j < NB; ++j) {
    const int n0 = j * 32;
    const __bf16* kbuf = ldsK[j & 1];
    const __bf16* vbuf = ldsV[j & 1];

    if (j + 1 < NB) {                            // stage next block, then wait
      stage_kv(Kb, Vb, n0 + 32, ldsK[(j + 1) & 1], ldsV[(j + 1) & 1], tid);
      WAIT_ASYNC_NEXT_INFLIGHT();                // drain current (oldest 8)
    } else {
      WAIT_ASYNC_ALL();
    }
    __syncthreads();                             // tiles visible block-wide

    // S = Q * K^T : two 16x16 tiles, K=128 in 4 chunks.
    // Issue all 16 ds_load_b128 first so DS returns overlap WMMA issue.
    v16bf kb[8];
#pragma unroll
    for (int kc = 0; kc < 4; ++kc) {
      kb[2 * kc]     = lds_b_frag(kbuf, 0,  kc * 32, KSTRIDE, lane);
      kb[2 * kc + 1] = lds_b_frag(kbuf, 16, kc * 32, KSTRIDE, lane);
    }
    v8f s0, s1;
#pragma unroll
    for (int r = 0; r < 8; ++r) { s0[r] = 0.f; s1[r] = 0.f; }
#pragma unroll
    for (int kc = 0; kc < 4; ++kc) {
      s0 = wmma_bf16(qa[kc], kb[2 * kc],     s0);
      s1 = wmma_bf16(qa[kc], kb[2 * kc + 1], s1);
    }

    // scale and per-row max (reduce across 16-lane half = key columns)
    float c0[8], c1[8], cmax[8];
#pragma unroll
    for (int r = 0; r < 8; ++r) {
      c0[r] = s0[r] * scale;
      c1[r] = s1[r] * scale;
      cmax[r] = fmaxf(c0[r], c1[r]);
    }
#pragma unroll
    for (int sh = 1; sh < 16; sh <<= 1)
#pragma unroll
      for (int r = 0; r < 8; ++r)
        cmax[r] = fmaxf(cmax[r], __shfl_xor(cmax[r], sh, 32));

    float cf[8], p0[8], p1[8], ps[8];
#pragma unroll
    for (int r = 0; r < 8; ++r) {
      float mn = fmaxf(mrow[r], cmax[r]);
      cf[r]   = exp2f((mrow[r] - mn) * LOG2E);
      mrow[r] = mn;
      p0[r] = exp2f((c0[r] - mn) * LOG2E);
      p1[r] = exp2f((c1[r] - mn) * LOG2E);
      ps[r] = p0[r] + p1[r];
    }
#pragma unroll
    for (int sh = 1; sh < 16; sh <<= 1)
#pragma unroll
      for (int r = 0; r < 8; ++r)
        ps[r] += __shfl_xor(ps[r], sh, 32);
#pragma unroll
    for (int r = 0; r < 8; ++r) lrow[r] = lrow[r] * cf[r] + ps[r];
#pragma unroll
    for (int t = 0; t < 8; ++t)
#pragma unroll
      for (int r = 0; r < 8; ++r) acc[t][r] *= cf[r];

    // repack P (C/D layout) -> A-matrix layout via per-wave LDS tile.
    // Same-wave DS ops execute in order; wave_barrier pins the schedule.
#pragma unroll
    for (int r = 0; r < 8; ++r) {
      pbuf[(r + 8 * hf) * 32 + mlane]      = (__bf16)p0[r];
      pbuf[(r + 8 * hf) * 32 + 16 + mlane] = (__bf16)p1[r];
    }
    __builtin_amdgcn_wave_barrier();
    union { v16bf v; v8bf h[2]; } pu;
    pu.h[0] = *(const v8bf*)(pbuf + mlane * 32 + hf * 8);
    pu.h[1] = *(const v8bf*)(pbuf + mlane * 32 + 16 + hf * 8);

    // context += P (16x32) * V (32 keys x 128 cv): prefetch all 8 B-frags,
    // then run the 8 WMMAs.
    v16bf vb[8];
#pragma unroll
    for (int t = 0; t < 8; ++t)
      vb[t] = lds_b_frag(vbuf, t * 16, 0, VSTRIDE, lane);
#pragma unroll
    for (int t = 0; t < 8; ++t)
      acc[t] = wmma_bf16(pu.v, vb[t], acc[t]);

    __syncthreads();   // everyone done with this buffer before it is re-staged
  }

  // normalize and write context (bf16, [B, HW, C_VAL])
  float inv[8];
#pragma unroll
  for (int r = 0; r < 8; ++r) inv[r] = 1.0f / lrow[r];
  __bf16* Cb = Ctx + (size_t)b * HWN * C_VAL;
#pragma unroll
  for (int t = 0; t < 8; ++t)
#pragma unroll
    for (int r = 0; r < 8; ++r) {
      int m  = m0 + r + 8 * hf;
      int cv = t * 16 + mlane;
      Cb[(size_t)m * C_VAL + cv] = (__bf16)(acc[t][r] * inv[r]);
    }
}

// ---------------------------------------------------------------------------
// Kernel 3: output projection out[b,co,hw] = sum_cv wo[co,cv]*ctx[b,hw,cv]+bo
// ---------------------------------------------------------------------------
__global__ __launch_bounds__(128) void proj_kernel(
    const __bf16* __restrict__ Ctx, const __bf16* __restrict__ wo_bf,
    const float* __restrict__ bo, float* __restrict__ out) {
  const int lane  = threadIdx.x & 31;
  const int wave  = threadIdx.x >> 5;
  const int b     = blockIdx.y;
  const int m0    = blockIdx.x * 64 + wave * 16;
  const int mlane = lane & 15;
  const int hf    = (lane >> 4) & 1;

  const __bf16* Cb = Ctx + (size_t)b * HWN * C_VAL;
  v16bf ca[4];
#pragma unroll
  for (int kc = 0; kc < 4; ++kc)
    ca[kc] = load_a_frag(Cb, m0, kc * 32, C_VAL, lane);

  float* ob = out + (size_t)b * C_OUT * HWN;
  for (int ct = 0; ct < 16; ++ct) {      // 16 output-channel tiles of 16
    v8f acc;
#pragma unroll
    for (int r = 0; r < 8; ++r) acc[r] = 0.f;
#pragma unroll
    for (int kc = 0; kc < 4; ++kc) {
      v16bf wb = load_b_frag(wo_bf, ct * 16, kc * 32, C_VAL, lane);
      acc = wmma_bf16(ca[kc], wb, acc);
    }
    int co = ct * 16 + mlane;
    float bias = bo[co];
#pragma unroll
    for (int r = 0; r < 8; ++r) {
      int m = m0 + r + 8 * hf;
      ob[(size_t)co * HWN + m] = acc[r] + bias;
    }
  }
}

// ---------------------------------------------------------------------------
extern "C" void kernel_launch(void* const* d_in, const int* in_sizes, int n_in,
                              void* d_out, int out_size, void* d_ws, size_t ws_size,
                              hipStream_t stream) {
  const float* x_enc = (const float*)d_in[0];
  const float* x_dec = (const float*)d_in[1];
  const float* wk    = (const float*)d_in[2];
  const float* bk    = (const float*)d_in[3];
  const float* gk    = (const float*)d_in[4];
  const float* betak = (const float*)d_in[5];
  const float* mk    = (const float*)d_in[6];
  const float* vk    = (const float*)d_in[7];
  const float* wq    = (const float*)d_in[8];
  const float* bq    = (const float*)d_in[9];
  const float* gq    = (const float*)d_in[10];
  const float* betaq = (const float*)d_in[11];
  const float* mq    = (const float*)d_in[12];
  const float* vq    = (const float*)d_in[13];
  const float* wv    = (const float*)d_in[14];
  const float* bv    = (const float*)d_in[15];
  const float* wo    = (const float*)d_in[16];
  const float* bo    = (const float*)d_in[17];
  float* out = (float*)d_out;

  const size_t QKELEMS = (size_t)BATCH * HWN * C_KEY;
  __bf16* Q     = (__bf16*)d_ws;
  __bf16* Kt    = Q  + QKELEMS;
  __bf16* Vt    = Kt + QKELEMS;
  __bf16* Ctx   = Vt + (size_t)BATCH * C_VAL * HWN;
  __bf16* wo_bf = Ctx + (size_t)BATCH * HWN * C_VAL;

  cvt_bf16_kernel<<<dim3((C_OUT * C_VAL + 255) / 256), 256, 0, stream>>>(
      wo, wo_bf, C_OUT * C_VAL);
  qkv_kernel<<<dim3(BATCH * HWN / 256, C_KEY, 3), 256, 0, stream>>>(
      x_enc, x_dec, wk, bk, gk, betak, mk, vk,
      wq, bq, gq, betaq, mq, vq, wv, bv, Q, Kt, Vt);
  attn_kernel<<<dim3(HWN / 64, BATCH), 128, 0, stream>>>(Q, Kt, Vt, Ctx);
  proj_kernel<<<dim3(HWN / 64, BATCH), 128, 0, stream>>>(Ctx, wo_bf, bo, out);
}
